// Head_59356448031312
// MI455X (gfx1250) — matrix operand order repaired and checked by reading, not compiled, not run on hardware
//
#include <hip/hip_runtime.h>

// ---------------------------------------------------------------------------
// Causal single-head attention (B=8, T=2048, C=768, H=64), MI455X / gfx1250.
// Phase 1: Q/K/V projections f32 -> bf16 via v_wmma_f32_16x16x32_bf16.
// Phase 2: flash attention (online softmax), bf16 WMMA, f32 accumulation,
//          V tiles staged global->LDS with the Tensor Data Mover
//          (tensor_load_to_lds + s_wait_tensorcnt).
// Workspace usage: 3 x (B*T*H) bf16 = 6 MiB.
// ---------------------------------------------------------------------------

typedef __attribute__((ext_vector_type(16))) __bf16 v16bf;
typedef __attribute__((ext_vector_type(8)))  float  v8f;

#define B_  8
#define T_  2048
#define C_  768
#define H_  64

// bf16 bit pattern of a float (RNE via hardware f32->bf16 convert)
__device__ __forceinline__ unsigned short bf16bits(float f) {
  union { __bf16 h; unsigned short u; } c; c.h = (__bf16)f; return c.u;
}

// A-matrix fragment (16x32 bf16), row-major [M][K] source.
// lane holds M = lane%16; element i -> K = kc + i + 16*hi.
__device__ __forceinline__ v16bf load_a16(const __bf16* row, int kc, int hi) {
  union { uint4 q[2]; v16bf v; } u;
  u.q[0] = *(const uint4*)(row + kc + 16 * hi);
  u.q[1] = *(const uint4*)(row + kc + 16 * hi + 8);
  return u.v;
}

// B-matrix fragment (32x16 bf16) from row-major [N][K] source (i.e. B^T rows).
// lane holds N = lane%16; element i -> K = off + i   (off = kc + 16*hi).
__device__ __forceinline__ v16bf load_b16(const __bf16* row, int off) {
  union { uint4 q[2]; v16bf v; } u;
  const uint4* p = (const uint4*)(row + off);
  u.q[0] = p[0];
  u.q[1] = p[1];
  return u.v;
}

__device__ __forceinline__ v8f wmma_bf16(v16bf a, v16bf b, v8f c) {
  // D = A x B + C, f32 accumulate
  return __builtin_amdgcn_wmma_f32_16x16x32_bf16(
      /*neg_a=*/false, a, /*neg_b=*/false, b,
      /*c_mod=*/(short)0, c, /*reuse_a=*/false, /*reuse_b=*/false);
}

// ---------------------------------------------------------------------------
// Phase 1: y = x @ W^T for W in {Wq, Wk, Wv}; x:[B*T,768] f32, W:[64,768] f32.
// blockIdx.y selects the projection. 8 waves/block, one 16-row x tile each.
// W tile (64 rows x 32 k) staged in LDS as bf16, padded stride 40 halves
// (80 B row stride -> 16 B aligned chunks, conflict-free column reads).
// ---------------------------------------------------------------------------
__global__ __launch_bounds__(256) void proj_qkv(
    const float* __restrict__ x,
    const float* __restrict__ Wk, const float* __restrict__ Wq,
    const float* __restrict__ Wv,
    __bf16* __restrict__ Qb, __bf16* __restrict__ Kb, __bf16* __restrict__ Vb)
{
  __shared__ unsigned short Wlds[64 * 40];

  const float* W; __bf16* dst;
  if      (blockIdx.y == 0) { W = Wq; dst = Qb; }
  else if (blockIdx.y == 1) { W = Wk; dst = Kb; }
  else                      { W = Wv; dst = Vb; }

  const int t    = threadIdx.x;
  const int wave = t >> 5;
  const int lane = t & 31;
  const int hi   = lane >> 4;
  const int lo   = lane & 15;
  const int m0   = (blockIdx.x * 8 + wave) * 16;

  const float* xrow = x + (size_t)(m0 + lo) * C_;

  const v8f vz = {0.f, 0.f, 0.f, 0.f, 0.f, 0.f, 0.f, 0.f};
  v8f acc[4] = {vz, vz, vz, vz};

  const int wrow = t >> 2;        // 0..63
  const int wcb  = (t & 3) * 8;   // 0,8,16,24

  for (int kc = 0; kc < C_; kc += 32) {
    // cooperative W tile -> LDS (f32 -> bf16)
    const float* wr = W + (size_t)wrow * C_ + kc + wcb;
    float4 f0 = *(const float4*)wr;
    float4 f1 = *(const float4*)(wr + 4);
    unsigned short* ws = Wlds + wrow * 40 + wcb;
    ws[0] = bf16bits(f0.x); ws[1] = bf16bits(f0.y);
    ws[2] = bf16bits(f0.z); ws[3] = bf16bits(f0.w);
    ws[4] = bf16bits(f1.x); ws[5] = bf16bits(f1.y);
    ws[6] = bf16bits(f1.z); ws[7] = bf16bits(f1.w);
    __syncthreads();

    // A fragment from x (f32 loads, convert in registers)
    float4 g0 = *(const float4*)(xrow + kc + 16 * hi);
    float4 g1 = *(const float4*)(xrow + kc + 16 * hi + 4);
    float4 g2 = *(const float4*)(xrow + kc + 16 * hi + 8);
    float4 g3 = *(const float4*)(xrow + kc + 16 * hi + 12);
    v16bf a;
    a[0]  = (__bf16)g0.x; a[1]  = (__bf16)g0.y; a[2]  = (__bf16)g0.z; a[3]  = (__bf16)g0.w;
    a[4]  = (__bf16)g1.x; a[5]  = (__bf16)g1.y; a[6]  = (__bf16)g1.z; a[7]  = (__bf16)g1.w;
    a[8]  = (__bf16)g2.x; a[9]  = (__bf16)g2.y; a[10] = (__bf16)g2.z; a[11] = (__bf16)g2.w;
    a[12] = (__bf16)g3.x; a[13] = (__bf16)g3.y; a[14] = (__bf16)g3.z; a[15] = (__bf16)g3.w;

    #pragma unroll
    for (int n = 0; n < 4; ++n) {
      v16bf bw = load_b16((const __bf16*)(Wlds + (n * 16 + lo) * 40), 16 * hi);
      acc[n] = wmma_bf16(a, bw, acc[n]);
    }
    __syncthreads();
  }

  // C/D layout: element r <-> (M = r + 8*hi, N = lane%16)
  #pragma unroll
  for (int n = 0; n < 4; ++n)
    #pragma unroll
    for (int r = 0; r < 8; ++r)
      dst[(size_t)(m0 + r + 8 * hi) * H_ + n * 16 + lo] = (__bf16)acc[n][r];
}

// ---------------------------------------------------------------------------
// Phase 2: flash attention. 4 independent waves per block; each wave owns one
// 16-query tile, streams 32-key blocks (K/V hot in L2: 256 KB/batch each).
// Per-wave LDS: P[16][40] (C-layout -> A-layout reshape), V^T[64][40],
// and Vstage[32][64] filled by the Tensor Data Mover.
// ---------------------------------------------------------------------------
__global__ __launch_bounds__(128) void attn_fwd(
    const __bf16* __restrict__ Qb, const __bf16* __restrict__ Kb,
    const __bf16* __restrict__ Vb, float* __restrict__ out)
{
  // per wave: 640 (P) + 2560 (V^T) + 2048 (Vstage) = 5248 ushorts (10.25 KB)
  __shared__ unsigned short smem[4 * 5248];   // 41 KB
  const int wave = threadIdx.x >> 5;
  const int lane = threadIdx.x & 31;
  const int hi   = lane >> 4;
  const int lo   = lane & 15;
  unsigned short* Psm = smem + wave * 5248;   // [16][40] bf16 bits
  unsigned short* VTs = Psm + 640;            // [64][40] bf16 bits (V^T)
  unsigned short* Vst = VTs + 2560;           // [32][64] bf16 bits (TDM dest)

  const int bb = blockIdx.y;
  const int q0 = (blockIdx.x * 4 + wave) * 16;
  const size_t base = (size_t)bb * T_;

  // LDS byte address of the TDM staging buffer (wave-uniform by construction;
  // readfirstlane makes that visible to the compiler so it can live in SGPRs).
  const unsigned vst_lds_addr = (unsigned)__builtin_amdgcn_readfirstlane(
      (int)(unsigned)(unsigned long long)(const void*)Vst);

  // Q tile: two A fragments covering head dim 0..31 and 32..63
  const __bf16* qrow = Qb + (base + q0 + lo) * H_;
  v16bf aq0 = load_a16(qrow, 0,  hi);
  v16bf aq1 = load_a16(qrow, 32, hi);

  const v8f vz = {0.f, 0.f, 0.f, 0.f, 0.f, 0.f, 0.f, 0.f};
  v8f o[4] = {vz, vz, vz, vz};
  float m_r[8], l_r[8];
  #pragma unroll
  for (int r = 0; r < 8; ++r) { m_r[r] = -3.0e38f; l_r[r] = 0.f; }

  const int nblk = (q0 + 47) >> 5;   // 32-key blocks covering keys 0..q0+15
  for (int jb = 0; jb < nblk; ++jb) {
    const int kb0 = jb * 32;

    // ---- TDM: async DMA of V block (32 rows x 128 B, contiguous 4 KB)
    // global -> LDS Vstage.  1-D descriptor: data_size=2 B, tile_dim0=2048.
    {
      const unsigned long long ga =
          (unsigned long long)(const void*)(Vb + (base + kb0) * H_);
      const unsigned g0w0 = 1u;                    // D# count = 1 (user)
      const unsigned g0w1 = vst_lds_addr;          // lds_addr [63:32]
      const unsigned g0w2 = (unsigned)ga;          // global_addr lo
      const unsigned g0w3 = (unsigned)((ga >> 32) & 0x01FFFFFFull)
                            | 0x80000000u;         // global_addr hi | type=2
      const unsigned g1w0 = 1u << 16;              // data_size = 1 (2 bytes)
      const unsigned g1w1 = 2048u << 16;           // tensor_dim0 = 2048 elems
      const unsigned g1w2 = 1u << 16;              // tensor_dim1 = 1
      const unsigned g1w3 = 2048u << 16;           // tile_dim0 = 2048 elems
      const unsigned g1w4 = 0u;                    // tile_dim1 = 0 (1-D)
      const unsigned g1w5 = 2048u;                 // tensor_dim0_stride
      const unsigned g1w6 = 0u;
      const unsigned g1w7 = 0u;
      // previous iteration's LDS reads of Vstage must drain before the DMA
      // may overwrite the staging buffer
      asm volatile("s_wait_dscnt 0x0" ::: "memory");
      asm volatile(
          "s_mov_b32 s64, %0\n\t"  "s_mov_b32 s65, %1\n\t"
          "s_mov_b32 s66, %2\n\t"  "s_mov_b32 s67, %3\n\t"
          "s_mov_b32 s68, %4\n\t"  "s_mov_b32 s69, %5\n\t"
          "s_mov_b32 s70, %6\n\t"  "s_mov_b32 s71, %7\n\t"
          "s_mov_b32 s72, %8\n\t"  "s_mov_b32 s73, %9\n\t"
          "s_mov_b32 s74, %10\n\t" "s_mov_b32 s75, %11\n\t"
          "tensor_load_to_lds s[64:67], s[68:75]"
          :
          : "s"(g0w0), "s"(g0w1), "s"(g0w2), "s"(g0w3),
            "s"(g1w0), "s"(g1w1), "s"(g1w2), "s"(g1w3),
            "s"(g1w4), "s"(g1w5), "s"(g1w6), "s"(g1w7)
          : "s64","s65","s66","s67","s68","s69","s70","s71",
            "s72","s73","s74","s75","memory");
    }

    if (jb + 1 < nblk)   // pull next block's K rows toward the cache
      __builtin_prefetch(Kb + (base + kb0 + 32 + lane) * H_, 0, 0);

    // S = Q K^T for two 16-key subtiles (K rows serve directly as B^T rows);
    // overlaps with the in-flight TDM transfer.
    const __bf16* kr0 = Kb + (base + kb0 + lo) * H_;
    const __bf16* kr1 = Kb + (base + kb0 + 16 + lo) * H_;
    v8f s0 = vz, s1 = vz;
    s0 = wmma_bf16(aq0, load_b16(kr0, 16 * hi),      s0);
    s0 = wmma_bf16(aq1, load_b16(kr0, 32 + 16 * hi), s0);
    s1 = wmma_bf16(aq0, load_b16(kr1, 16 * hi),      s1);
    s1 = wmma_bf16(aq1, load_b16(kr1, 32 + 16 * hi), s1);

    // Wait for the TDM transfer, then transpose Vstage -> V^T in LDS
    __builtin_amdgcn_s_wait_tensorcnt(0);
    {
      const uint4* vr = (const uint4*)(Vst + lane * 64);  // lane owns one V row
      #pragma unroll
      for (int i = 0; i < 8; ++i) {
        uint4 d = vr[i];
        unsigned int wv[4] = {d.x, d.y, d.z, d.w};
        #pragma unroll
        for (int j = 0; j < 4; ++j) {
          int h = i * 8 + j * 2;
          VTs[h * 40 + lane]       = (unsigned short)(wv[j] & 0xffffu);
          VTs[(h + 1) * 40 + lane] = (unsigned short)(wv[j] >> 16);
        }
      }
    }

    // Causal mask + online softmax (rows live in 16-lane groups)
    float corr[8];
    #pragma unroll
    for (int r = 0; r < 8; ++r) {
      const int qi = q0 + r + 8 * hi;
      if (kb0 + lo > qi)      s0[r] = -1.0e30f;
      if (kb0 + 16 + lo > qi) s1[r] = -1.0e30f;
      float mx = fmaxf(s0[r], s1[r]);
      mx = fmaxf(mx, __shfl_xor(mx, 1, 32));
      mx = fmaxf(mx, __shfl_xor(mx, 2, 32));
      mx = fmaxf(mx, __shfl_xor(mx, 4, 32));
      mx = fmaxf(mx, __shfl_xor(mx, 8, 32));
      const float mn = fmaxf(m_r[r], mx);
      const float cf = __expf(m_r[r] - mn);
      const float p0 = __expf(s0[r] - mn);
      const float p1 = __expf(s1[r] - mn);
      float rs = p0 + p1;
      rs += __shfl_xor(rs, 1, 32);
      rs += __shfl_xor(rs, 2, 32);
      rs += __shfl_xor(rs, 4, 32);
      rs += __shfl_xor(rs, 8, 32);
      l_r[r] = l_r[r] * cf + rs;
      m_r[r] = mn;
      corr[r] = cf;
      // P in LDS: row = query-local M, col = key-local
      Psm[(r + 8 * hi) * 40 + lo]      = bf16bits(p0);
      Psm[(r + 8 * hi) * 40 + 16 + lo] = bf16bits(p1);
    }
    #pragma unroll
    for (int n = 0; n < 4; ++n)
      #pragma unroll
      for (int r = 0; r < 8; ++r)
        o[n][r] *= corr[r];

    // O += P V : A from P (reshaped via LDS), B from V^T (contiguous 32B)
    v16bf ap = load_a16((const __bf16*)(Psm + lo * 40), 0, hi);
    #pragma unroll
    for (int n = 0; n < 4; ++n) {
      v16bf bv = load_b16((const __bf16*)(VTs + (n * 16 + lo) * 40), 16 * hi);
      o[n] = wmma_bf16(ap, bv, o[n]);
    }
  }

  // Normalize and write f32 output [B,T,H]
  #pragma unroll
  for (int r = 0; r < 8; ++r) {
    const float inv = 1.0f / l_r[r];
    const size_t rowoff = (base + q0 + r + 8 * hi) * H_;
    #pragma unroll
    for (int n = 0; n < 4; ++n)
      out[rowoff + n * 16 + lo] = o[n][r] * inv;
  }
}

// ---------------------------------------------------------------------------
extern "C" void kernel_launch(void* const* d_in, const int* in_sizes, int n_in,
                              void* d_out, int out_size, void* d_ws, size_t ws_size,
                              hipStream_t stream) {
  const float* x  = (const float*)d_in[0];
  const float* Wk = (const float*)d_in[1];
  const float* Wq = (const float*)d_in[2];
  const float* Wv = (const float*)d_in[3];
  float* out = (float*)d_out;

  // Workspace: bf16 Q, K, V  (each B_*T_*H_ = 1,048,576 elems = 2 MiB)
  __bf16* qb = (__bf16*)d_ws;
  __bf16* kb = qb + (size_t)B_ * T_ * H_;
  __bf16* vb = kb + (size_t)B_ * T_ * H_;

  // Phase 1: projections. grid.x covers B*T/(16 rows * 8 waves) = 128 blocks.
  proj_qkv<<<dim3(128, 3), 256, 0, stream>>>(x, Wk, Wq, Wv, qb, kb, vb);

  // Phase 2: flash attention. 4 q-tiles (waves) per block: T/(16*4)=32 blocks/batch.
  attn_fwd<<<dim3(T_ / 64, B_), 128, 0, stream>>>(qb, kb, vb, out);
}
